// CrossSparseGAT_44169443672637
// MI455X (gfx1250) — matrix operand order repaired and checked by reading, not compiled, not run on hardware
//
#include <hip/hip_runtime.h>
#include <hip/hip_bf16.h>

typedef __attribute__((ext_vector_type(16))) _Float16 v16h;
typedef __attribute__((ext_vector_type(8)))  _Float16 v8h;
typedef __attribute__((ext_vector_type(8)))  float    v8f;
typedef __attribute__((ext_vector_type(4)))  unsigned u32x4;
typedef __attribute__((ext_vector_type(8)))  int      i32x8;
typedef __attribute__((ext_vector_type(4)))  int      i32x4;

#define N_DST 50000
#define N_SRC 50000
#define E_NUM 500000
#define DIM   128
#define NH    8
#define HD    16

// ---------------- ordered-float encoding for atomic max ----------------
__device__ __forceinline__ unsigned enc_f(float x) {
    unsigned u = __float_as_uint(x);
    return (u & 0x80000000u) ? ~u : (u | 0x80000000u);
}
__device__ __forceinline__ float dec_f(unsigned k) {
    return (k & 0x80000000u) ? __uint_as_float(k ^ 0x80000000u)
                             : __uint_as_float(~k);
}

// ---------------- WMMA fragment helpers (16x16x32 f16) ----------------
// A fragment: 16x32 (MxK) f16 tile stored row-major in LDS (row stride 128).
// ISA layout: lanes0-15 -> M=lane, K{0..7,16..23}; lanes16-31 -> M=lane-16, K{8..15,24..31}
__device__ __forceinline__ v16h load_a_frag(const _Float16* As, int lane, int k0) {
    int m  = lane & 15;
    int hi = lane >> 4;
    const _Float16* p = As + m * DIM + k0 + hi * 8;
    v8h lo = *(const v8h*)(p);
    v8h hh = *(const v8h*)(p + 16);
    v16h a;
#pragma unroll
    for (int j = 0; j < 8; ++j) { a[j] = lo[j]; a[j + 8] = hh[j]; }
    return a;
}

// Packed-B fragment: weights are pre-swizzled so each lane's 16 halves are
// contiguous (32B). Index: ((ks*8 + ntile)*32 + lane)*16.
__device__ __forceinline__ v16h load_b_packed(const _Float16* Bp, int lane, int ks, int ntile) {
    return *(const v16h*)(Bp + (size_t)(((ks * 8) + ntile) * 32 + lane) * 16);
}

// ---------------- K0: tiny weight precompute ----------------
// Wc1 = W1@W4 [128,8], Wc2 = W2@W4 [128,8], w3w4 = W3@W4 [8]
__global__ void prep_weights(const float* __restrict__ W1, const float* __restrict__ W2,
                             const float* __restrict__ W3, const float* __restrict__ W4,
                             float* __restrict__ Wc1, float* __restrict__ Wc2,
                             float* __restrict__ w3w4) {
    int t = threadIdx.x;
    for (int p = t; p < DIM * NH; p += blockDim.x) {
        int i = p >> 3, h = p & 7;
        float a1 = 0.f, a2 = 0.f;
        for (int j = 0; j < DIM; ++j) {
            float w4 = W4[j * NH + h];
            a1 += W1[i * DIM + j] * w4;
            a2 += W2[i * DIM + j] * w4;
        }
        Wc1[p] = a1; Wc2[p] = a2;
    }
    if (t < NH) {
        float a = 0.f;
        for (int j = 0; j < DIM; ++j) a += W3[j] * W4[j * NH + t];
        w3w4[t] = a;
    }
}

// ---------------- K0b: pack a [128,128] f32 weight into f16 B-fragment layout ----------------
__global__ void pack_weight(const float* __restrict__ W, _Float16* __restrict__ out) {
    int gid = blockIdx.x * blockDim.x + threadIdx.x;   // 0..16383
    if (gid >= 4 * 8 * 32 * 16) return;
    int j    = gid & 15;
    int lane = (gid >> 4) & 31;
    int nt   = (gid >> 9) & 7;
    int ks   = gid >> 12;
    int k = ks * 32 + ((j >> 3) << 4) + ((lane >> 4) << 3) + (j & 7);
    int n = nt * 16 + (lane & 15);
    out[gid] = (_Float16)W[k * DIM + n];
}

// ---------------- K1: per-node per-head scores a = feats @ Wc ----------------
__global__ void node_head(const float* __restrict__ feats, const float* __restrict__ Wc,
                          float* __restrict__ out, int total) {
    int gid = blockIdx.x * blockDim.x + threadIdx.x;
    if (gid >= total) return;
    int n = gid >> 3, h = gid & 7;
    const float* fr = feats + (size_t)n * DIM;
    float acc = 0.f;
#pragma unroll 8
    for (int k = 0; k < DIM; ++k) acc += fr[k] * Wc[k * NH + h];
    out[gid] = acc;
}

// ---------------- K2: WMMA GEMM, 64 rows/block, TDM-staged A tile ----------------
// C[64,128] = A[64,128] @ B[128,128]  (B pre-packed f16)
__global__ void gemm64(const float* __restrict__ A, const _Float16* __restrict__ Bp,
                       float* __restrict__ C, int nrows) {
    __shared__ __align__(16) float    As32[64 * DIM];
    __shared__ __align__(16) _Float16 Ash[64 * DIM];
    int row0 = blockIdx.x * 64;

    if (threadIdx.x < 32) {
        // Build Tensor DMA descriptor: 2-D tile 128 x 64 of 4-byte elements.
        unsigned lds_off = (unsigned)(size_t)(void*)As32;   // low 32 bits = LDS byte offset
        unsigned long long ga = (unsigned long long)(size_t)(const void*)(A + (size_t)row0 * DIM);
        int rem = nrows - row0;                             // rows left; OOB rows read as 0
        u32x4 g0;
        g0[0] = 1u;                                         // count=1 (valid), user mode
        g0[1] = lds_off;                                    // lds_addr
        g0[2] = (unsigned)(ga & 0xFFFFFFFFull);             // global_addr[31:0]
        g0[3] = (unsigned)((ga >> 32) & 0x01FFFFFFull)      // global_addr[56:32]
              | 0x80000000u;                                // type=2 ("image")
        i32x8 g1;
        g1[0] = 0x00020000;                                 // data_size=2 (4B), no mask/pad
        g1[1] = (int)(DIM << 16);                           // tensor_dim0 = 128
        g1[2] = (int)((rem & 0xFFFF) << 16);                // tensor_dim1[15:0]
        g1[3] = (int)(((unsigned)rem >> 16) & 0xFFFFu)      // tensor_dim1[31:16]
              | (DIM << 16);                                // tile_dim0 = 128
        g1[4] = 64;                                         // tile_dim1 = 64
        g1[5] = DIM;                                        // tensor_dim0_stride = 128
        g1[6] = 0;
        g1[7] = 0;
        i32x4 z4 = {};
#if defined(__clang_major__) && (__clang_major__ >= 23)
        i32x8 z8 = {};
        __builtin_amdgcn_tensor_load_to_lds(g0, g1, z4, z4, z8, 0);
#else
        __builtin_amdgcn_tensor_load_to_lds(g0, g1, z4, z4, 0);
#endif
        __builtin_amdgcn_s_wait_tensorcnt((short)0);
    }
    __syncthreads();

    // f32 -> f16 for WMMA fragments
    for (int p = threadIdx.x; p < 64 * DIM; p += blockDim.x)
        Ash[p] = (_Float16)As32[p];
    __syncthreads();

    int lane = threadIdx.x & 31;
    int wave = threadIdx.x >> 5;       // 8 waves -> 8 column tiles
    v8f acc[4] = {v8f{}, v8f{}, v8f{}, v8f{}};
#pragma unroll
    for (int ks = 0; ks < 4; ++ks) {
        v16h b = load_b_packed(Bp, lane, ks, wave);
#pragma unroll
        for (int rt = 0; rt < 4; ++rt) {
            v16h a = load_a_frag(Ash + rt * 16 * DIM, lane, ks * 32);
            acc[rt] = __builtin_amdgcn_wmma_f32_16x16x32_f16(false, a, false, b,
                                                             (short)0, acc[rt], false, false);
        }
    }
    int n = lane & 15, hi = lane >> 4, n0 = wave * 16;
#pragma unroll
    for (int rt = 0; rt < 4; ++rt) {
#pragma unroll
        for (int r = 0; r < 8; ++r) {
            int row = row0 + rt * 16 + r + hi * 8;
            if (row < nrows)
                C[(size_t)row * DIM + n0 + n] = acc[rt][r];
        }
    }
}

// ---------------- K3: zero scratch (mkey | ssum | agg contiguous) ----------------
__global__ void zero_ws(float* __restrict__ p, int n) {
    int gid = blockIdx.x * blockDim.x + threadIdx.x;
    if (gid < n) p[gid] = 0.0f;   // enc-key 0 == below every real encoded float
}

// ---------------- K4: edge pass 1, segment max of leaky-relu logits ----------------
__global__ void edge_max(const long long* __restrict__ ei, const float* __restrict__ P,
                         const float* __restrict__ det,
                         const float* __restrict__ a_dst, const float* __restrict__ a_src,
                         const float* __restrict__ w3w4, unsigned* __restrict__ mkey) {
    int gid = blockIdx.x * blockDim.x + threadIdx.x;
    if (gid >= E_NUM * NH) return;
    int e = gid >> 3, h = gid & 7;
    long long s = ei[e];
    long long d = ei[E_NUM + e];
    float x = a_dst[d * NH + h] + a_src[s * NH + h] + P[e] * w3w4[h] + det[e];
    x = (x < 0.f) ? 0.2f * x : x;
    atomicMax(&mkey[d * NH + h], enc_f(x));
}

// ---------------- K5: edge pass 2, exp / ssum / scatter exp*V ----------------
__global__ void edge_accum(const long long* __restrict__ ei, const float* __restrict__ P,
                           const float* __restrict__ det,
                           const float* __restrict__ a_dst, const float* __restrict__ a_src,
                           const float* __restrict__ w3w4, const unsigned* __restrict__ mkey,
                           float* __restrict__ ssum, float* __restrict__ agg,
                           const float* __restrict__ V) {
    int gid = blockIdx.x * blockDim.x + threadIdx.x;
    if (gid >= E_NUM * DIM) return;
    int e = gid >> 7, c = gid & 127, h = c >> 4;
    long long s = ei[e];
    long long d = ei[E_NUM + e];
    float x = a_dst[d * NH + h] + a_src[s * NH + h] + P[e] * w3w4[h] + det[e];
    x = (x < 0.f) ? 0.2f * x : x;
    float m  = dec_f(mkey[d * NH + h]);
    float ex = __expf(x - m);
    if ((c & 15) == 0) atomicAdd(&ssum[d * NH + h], ex);
    atomicAdd(&agg[d * DIM + c], ex * V[s * DIM + c]);
}

// ---------------- K6: fused  out = LN( norm(agg)@Wout + b + dst@res_w + rb ) ----------------
__global__ void final_fused(const float* __restrict__ agg, const float* __restrict__ ssum,
                            const float* __restrict__ dst_feats,
                            const _Float16* __restrict__ WpOut, const float* __restrict__ Wout_b,
                            const _Float16* __restrict__ WpRes, const float* __restrict__ res_b,
                            const float* __restrict__ ln_g,     const float* __restrict__ ln_b,
                            float* __restrict__ out) {
    __shared__ __align__(16) _Float16 A1[16 * DIM];   // normalized agg
    __shared__ __align__(16) _Float16 A2[16 * DIM];   // dst_feats
    __shared__ float Ot[16 * DIM];
    __shared__ float redS[16 * 16], redQ[16 * 16];
    __shared__ float muS[16], rsS[16];

    int row0 = blockIdx.x * 16;
    for (int p = threadIdx.x; p < 16 * DIM; p += blockDim.x) {
        int m = p >> 7, k = p & 127, h = k >> 4;
        int row = row0 + m;
        float inv = 1.0f / (ssum[row * NH + h] + 1e-12f);
        A1[p] = (_Float16)(agg[(size_t)row * DIM + k] * inv);
        A2[p] = (_Float16)dst_feats[(size_t)row * DIM + k];
    }
    __syncthreads();

    int lane = threadIdx.x & 31;
    int wave = threadIdx.x >> 5;
    int n0 = wave * 16;
    v8f acc = {};
#pragma unroll
    for (int ks = 0; ks < 4; ++ks) {
        v16h a = load_a_frag(A1, lane, ks * 32);
        v16h b = load_b_packed(WpOut, lane, ks, wave);
        acc = __builtin_amdgcn_wmma_f32_16x16x32_f16(false, a, false, b,
                                                     (short)0, acc, false, false);
    }
#pragma unroll
    for (int ks = 0; ks < 4; ++ks) {
        v16h a = load_a_frag(A2, lane, ks * 32);
        v16h b = load_b_packed(WpRes, lane, ks, wave);
        acc = __builtin_amdgcn_wmma_f32_16x16x32_f16(false, a, false, b,
                                                     (short)0, acc, false, false);
    }
    int n = lane & 15, hi = lane >> 4;
    float bias = Wout_b[n0 + n] + res_b[n0 + n];
#pragma unroll
    for (int r = 0; r < 8; ++r) {
        int row = r + hi * 8;
        Ot[row * DIM + n0 + n] = acc[r] + bias;
    }
    __syncthreads();

    // LayerNorm over each 128-wide row: 16 threads per row, 8 elems each
    int r  = threadIdx.x >> 4;
    int sg = threadIdx.x & 15;
    float s = 0.f, q = 0.f;
#pragma unroll
    for (int i = 0; i < 8; ++i) {
        float v = Ot[r * DIM + sg * 8 + i];
        s += v; q += v * v;
    }
    redS[r * 16 + sg] = s; redQ[r * 16 + sg] = q;
    __syncthreads();
    if (threadIdx.x < 16) {
        float ts = 0.f, tq = 0.f;
        for (int i = 0; i < 16; ++i) { ts += redS[threadIdx.x * 16 + i]; tq += redQ[threadIdx.x * 16 + i]; }
        float mu  = ts * (1.0f / DIM);
        float var = tq * (1.0f / DIM) - mu * mu;
        muS[threadIdx.x] = mu;
        rsS[threadIdx.x] = rsqrtf(var + 1e-5f);
    }
    __syncthreads();
    float mu = muS[r], rs = rsS[r];
#pragma unroll
    for (int i = 0; i < 8; ++i) {
        int k = sg * 8 + i;
        float v = Ot[r * DIM + k];
        out[(size_t)(row0 + r) * DIM + k] = (v - mu) * rs * ln_g[k] + ln_b[k];
    }
}

extern "C" void kernel_launch(void* const* d_in, const int* in_sizes, int n_in,
                              void* d_out, int out_size, void* d_ws, size_t ws_size,
                              hipStream_t stream) {
    const float*     dst_feats = (const float*)d_in[0];
    const float*     src_feats = (const float*)d_in[1];
    const long long* ei        = (const long long*)d_in[2];
    const float*     P         = (const float*)d_in[3];
    const float*     det       = (const float*)d_in[4];
    const float*     W1        = (const float*)d_in[5];
    const float*     W2        = (const float*)d_in[6];
    const float*     W3        = (const float*)d_in[7];
    const float*     W4        = (const float*)d_in[8];
    const float*     Wv        = (const float*)d_in[9];
    const float*     Wout_w    = (const float*)d_in[10];
    const float*     Wout_b    = (const float*)d_in[11];
    const float*     res_w     = (const float*)d_in[12];
    const float*     res_b     = (const float*)d_in[13];
    const float*     ln_g      = (const float*)d_in[14];
    const float*     ln_b      = (const float*)d_in[15];

    float* ws = (float*)d_ws;
    float*    Wc1   = ws;                                  // 1024
    float*    Wc2   = ws + 1024;                           // 1024
    float*    w3w4  = ws + 2048;                           // 8 (padded)
    float*    a_dst = ws + 4096;                           // 400000
    float*    a_src = a_dst + (size_t)N_DST * NH;          // 400000
    float*    V     = a_src + (size_t)N_SRC * NH;          // 6,400,000
    unsigned* mkey  = (unsigned*)(V + (size_t)N_SRC * DIM);// 400000
    float*    ssum  = (float*)(mkey + (size_t)N_DST * NH); // 400000
    float*    agg   = ssum + (size_t)N_DST * NH;           // 6,400,000
    _Float16* WpV   = (_Float16*)(agg + (size_t)N_DST * DIM); // 16384 halves each
    _Float16* WpOut = WpV + 16384;
    _Float16* WpRes = WpOut + 16384;

    prep_weights<<<1, 256, 0, stream>>>(W1, W2, W3, W4, Wc1, Wc2, w3w4);
    pack_weight<<<64, 256, 0, stream>>>(Wv,     WpV);
    pack_weight<<<64, 256, 0, stream>>>(Wout_w, WpOut);
    pack_weight<<<64, 256, 0, stream>>>(res_w,  WpRes);

    int nh_total = N_DST * NH;
    node_head<<<(nh_total + 255) / 256, 256, 0, stream>>>(dst_feats, Wc1, a_dst, nh_total);
    node_head<<<(nh_total + 255) / 256, 256, 0, stream>>>(src_feats, Wc2, a_src, nh_total);

    gemm64<<<(N_SRC + 63) / 64, 256, 0, stream>>>(src_feats, WpV, V, N_SRC);

    int zero_n = N_DST * NH * 2 + N_DST * DIM;             // mkey + ssum + agg (contiguous)
    zero_ws<<<(zero_n + 255) / 256, 256, 0, stream>>>((float*)mkey, zero_n);

    edge_max<<<(E_NUM * NH + 255) / 256, 256, 0, stream>>>(ei, P, det, a_dst, a_src, w3w4, mkey);

    edge_accum<<<(E_NUM * DIM + 255) / 256, 256, 0, stream>>>(ei, P, det, a_dst, a_src,
                                                              w3w4, mkey, ssum, agg, V);

    final_fused<<<N_DST / 16, 256, 0, stream>>>(agg, ssum, dst_feats, WpOut, Wout_b,
                                                WpRes, res_b, ln_g, ln_b, (float*)d_out);
}
